// SelfAttention_36550171689004
// MI455X (gfx1250) — matrix operand order repaired and checked by reading, not compiled, not run on hardware
//
#include <hip/hip_runtime.h>

// Self-attention: B=2, S=2048, D=1024, H=16, hd=64. All matmuls via
// v_wmma_f32_16x16x32_f16 (f16 inputs, fp32 accumulate).
// v4: double-buffered LDS staging fed by GLOBAL_LOAD_ASYNC_TO_LDS_B128
// (ASYNCcnt-tracked, no VGPR round-trip); next tile's transfer overlaps
// the current tile's WMMAs. Workgroup split-barriers between phases.

typedef __attribute__((ext_vector_type(16))) _Float16 v16h;
typedef __attribute__((ext_vector_type(8)))  _Float16 v8h;
typedef __attribute__((ext_vector_type(8)))  float    v8f;

constexpr int Bc  = 2;
constexpr int Sc  = 2048;
constexpr int Dc  = 1024;
constexpr int Hc  = 16;
constexpr int HDc = 64;
constexpr int Mrows = Bc * Sc;   // 4096
constexpr int Nqkv  = 3 * Dc;    // 3072

#define LOG2E 1.4426950408889634f

union V16U { v16h v; v8h h[2]; };

// ---- CDNA5 async global->LDS staging (CDNA5 ISA) ----

// 16 bytes per lane, memory -> LDS directly, tracked by ASYNCcnt.
__device__ __forceinline__ void async_load_b128(const void* gaddr, void* lds) {
  unsigned lds_off = (unsigned)(unsigned long long)lds;   // generic low 32b = LDS offset
  asm volatile("global_load_async_to_lds_b128 %0, %1, off"
               :: "v"(lds_off), "v"((unsigned long long)gaddr)
               : "memory");
}

__device__ __forceinline__ void wait_async0() {
  asm volatile("s_wait_asynccnt 0" ::: "memory");
}

// ---- WMMA fragment loaders (contiguous k = 16*(lane/16) + s, same order in A and B) ----

// A-matrix 16x32 f16: lane holds one row; lanes 0-15 -> K0..15,
// lanes 16-31 -> K16..31. Two contiguous 16B loads.
__device__ __forceinline__ v16h frag_a_load(const _Float16* A, int lda, int lane) {
  int row   = lane & 15;
  int kbase = (lane < 16) ? 0 : 16;
  const _Float16* p = A + row * lda + kbase;
  V16U u;
  u.h[0] = *(const v8h*)p;
  u.h[1] = *(const v8h*)(p + 8);
  return u.v;
}

// B-matrix 32x16 f16 supplied as B^T row-major [N x K]: lane holds one
// column; lanes 0-15 -> K0..15, lanes 16-31 -> K16..31. One 32B load.
__device__ __forceinline__ v16h frag_b_load(const _Float16* Bt, int ldb, int lane) {
  int col  = lane & 15;
  int kofs = (lane < 16) ? 0 : 16;
  return *(const v16h*)(Bt + col * ldb + kofs);
}

#define WMMA_F16(a, b, c) \
  __builtin_amdgcn_wmma_f32_16x16x32_f16(false, (a), false, (b), (short)0, (c), false, false)

// ---------------------------------------------------------------------------
__global__ void f32_to_f16_kernel(const float* __restrict__ src,
                                  _Float16* __restrict__ dst, int n) {
  int i = blockIdx.x * blockDim.x + threadIdx.x;
  int stride = gridDim.x * blockDim.x;
  for (; i < n; i += stride) dst[i] = (_Float16)src[i];
}

// Block-cooperative 128x128 GEMM body. 8 waves in 4(M) x 2(N); each wave
// accumulates a 32x64 sub-tile. A/B 128x32 k-slices double-buffered in LDS,
// filled by async loads; slice k+1 prefetched under slice k's WMMAs.
template <int KDIM>
__device__ __forceinline__ void gemm_block_128x128(
    const _Float16* __restrict__ A,   // global: row m0_block, k 0
    const _Float16* __restrict__ Bt,  // global: row n0_block, k 0
    _Float16* As, _Float16* Bs,       // LDS, 2 x 128*32 f16 each (16 KB)
    int lane, int wm, int wn, v8f acc[2][4]) {
  constexpr int SLICE = 128 * 32;
  const int tid  = threadIdx.x;       // 0..255
  const int srow = tid >> 1;          // 0..127
  const int scol = (tid & 1) * 16;    // 0 | 16
  const int soff = srow * 32 + scol;

  const _Float16* ga = A  + (size_t)srow * KDIM + scol;
  const _Float16* gb = Bt + (size_t)srow * KDIM + scol;

  // prologue: stage k-slice 0 into buffer 0
  async_load_b128(ga, As + soff);
  async_load_b128(gb, Bs + soff);
  wait_async0();
  __syncthreads();

  for (int k0 = 0; k0 < KDIM; k0 += 32) {
    const int cur = (k0 >> 5) & 1;
    if (k0 + 32 < KDIM) {           // prefetch next slice into other buffer
      const int nxt = cur ^ 1;
      async_load_b128(ga + k0 + 32, As + nxt * SLICE + soff);
      async_load_b128(gb + k0 + 32, Bs + nxt * SLICE + soff);
    }

    const _Float16* Aw = As + cur * SLICE + (wm * 32) * 32;
    const _Float16* Bw = Bs + cur * SLICE + (wn * 64) * 32;
    v16h a0 = frag_a_load(Aw,           32, lane);
    v16h a1 = frag_a_load(Aw + 16 * 32, 32, lane);
    v16h b0 = frag_b_load(Bw,           32, lane);
    v16h b1 = frag_b_load(Bw + 16 * 32, 32, lane);
    v16h b2 = frag_b_load(Bw + 32 * 32, 32, lane);
    v16h b3 = frag_b_load(Bw + 48 * 32, 32, lane);
    acc[0][0] = WMMA_F16(a0, b0, acc[0][0]);
    acc[0][1] = WMMA_F16(a0, b1, acc[0][1]);
    acc[0][2] = WMMA_F16(a0, b2, acc[0][2]);
    acc[0][3] = WMMA_F16(a0, b3, acc[0][3]);
    acc[1][0] = WMMA_F16(a1, b0, acc[1][0]);
    acc[1][1] = WMMA_F16(a1, b1, acc[1][1]);
    acc[1][2] = WMMA_F16(a1, b2, acc[1][2]);
    acc[1][3] = WMMA_F16(a1, b3, acc[1][3]);

    wait_async0();     // next slice landed (overlapped with WMMAs above)
    __syncthreads();   // all waves done reading cur, staging visible
  }
}

// QKV projection: X[4096x1024] @ Wqkv^T. Per-wave 64-aligned col block lies
// entirely inside the q, k, or v region of one head -> uniform routing.
__global__ __launch_bounds__(256) void qkv_gemm_kernel(
    const _Float16* __restrict__ X,
    const _Float16* __restrict__ W,
    _Float16* __restrict__ Q,
    _Float16* __restrict__ K,
    _Float16* __restrict__ Vt) {
  __shared__ __align__(32) _Float16 As[2 * 128 * 32];
  __shared__ __align__(32) _Float16 Bs[2 * 128 * 32];

  const int lane = threadIdx.x & 31;
  const int wave = threadIdx.x >> 5;
  const int wm = wave >> 1, wn = wave & 1;
  const int nblk = Nqkv / 128;                   // 24
  const int mb = blockIdx.x / nblk, nb = blockIdx.x % nblk;
  const int m0 = mb * 128 + wm * 32;
  const int n0 = nb * 128 + wn * 64;

  v8f acc[2][4] = {};
  gemm_block_128x128<Dc>(X + (size_t)(mb * 128) * Dc, W + (size_t)(nb * 128) * Dc,
                         As, Bs, lane, wm, wn, acc);

  const int h      = n0 / (3 * HDc);             // head (uniform)
  const int region = (n0 % (3 * HDc)) / HDc;     // 0=q 1=k 2=v (uniform)
  const int hi = (lane >> 4) << 3;
  const int ln = lane & 15;

  #pragma unroll
  for (int i = 0; i < 2; ++i) {
    #pragma unroll
    for (int r = 0; r < 8; ++r) {
      int row = m0 + i * 16 + r + hi;
      int bb = row >> 11;                        // / Sc
      int s  = row & (Sc - 1);
      size_t base  = ((size_t)(bb * Hc + h) * Sc + s) * HDc;
      size_t vbase = ((size_t)(bb * Hc + h) * HDc) * Sc + s;
      #pragma unroll
      for (int j = 0; j < 4; ++j) {
        int d = j * 16 + ln;
        float val = acc[i][j][r];
        if (region == 0) {
          Q[base + d] = (_Float16)(val * 0.125f);
        } else if (region == 1) {
          K[base + d] = (_Float16)val;
        } else {
          Vt[vbase + (size_t)d * Sc] = (_Float16)val;
        }
      }
    }
  }
}

// Flash attention: block = 8 waves sharing one (b,h), 256 consecutive
// queries (32 per wave). K/V tiles double-buffered in LDS via async loads;
// tile j+32 prefetched under tile j's WMMAs + softmax.
__global__ __launch_bounds__(256) void attn_kernel(
    const _Float16* __restrict__ Q,
    const _Float16* __restrict__ K,
    const _Float16* __restrict__ Vt,
    _Float16* __restrict__ Oout) {
  __shared__ __align__(32) _Float16 Ks[2][32 * 64];     //  8 KB [key][dim]
  __shared__ __align__(32) _Float16 Vs[2][64 * 32];     //  8 KB [dim][key]
  __shared__ __align__(32) _Float16 Plds[8][32 * 32];   // 16 KB

  const int tid  = threadIdx.x;
  const int lane = tid & 31;
  const int wave = tid >> 5;
  const int qsegs = Sc / 256;                    // 8
  const int bh   = blockIdx.x / qsegs;
  const int qseg = blockIdx.x % qsegs;
  const int b = bh / Hc, h = bh % Hc;
  const int q0 = qseg * 256 + wave * 32;

  const _Float16* Qbase = Q  + (size_t)(b * Hc + h) * Sc * HDc;
  const _Float16* Kbase = K  + (size_t)(b * Hc + h) * Sc * HDc;
  const _Float16* Vbase = Vt + (size_t)(b * Hc + h) * HDc * Sc;

  v16h qf[2][2];
  #pragma unroll
  for (int qi = 0; qi < 2; ++qi)
    #pragma unroll
    for (int kk = 0; kk < 2; ++kk)
      qf[qi][kk] = frag_a_load(Qbase + (q0 + qi * 16) * HDc + kk * 32, HDc, lane);

  v8f o[2][4] = {};
  float mrow[2][8], lrow[2][8];
  #pragma unroll
  for (int qi = 0; qi < 2; ++qi)
    #pragma unroll
    for (int r = 0; r < 8; ++r) { mrow[qi][r] = -3.0e38f; lrow[qi][r] = 0.0f; }

  _Float16* P = Plds[wave];
  const int hi = (lane >> 4) << 3;
  const int ln = lane & 15;

  // staging: K tile 32x64 = 4 KB (16B/thread), V tile 64x32 likewise
  const int krow = tid >> 3, kseg = (tid & 7) * 8;   // K: [krow][kseg..+7]
  const int vrow = tid >> 2, vseg = (tid & 3) * 8;   // V: [vrow][vseg..+7]
  const _Float16* gK = Kbase + (size_t)krow * HDc + kseg;  // + j0*HDc
  const _Float16* gV = Vbase + (size_t)vrow * Sc + vseg;   // + j0
  _Float16* sK0 = &Ks[0][krow * 64 + kseg];
  _Float16* sV0 = &Vs[0][vrow * 32 + vseg];
  constexpr int KSL = 32 * 64, VSL = 64 * 32;

  // prologue: stage key tile 0 into buffer 0
  async_load_b128(gK, sK0);
  async_load_b128(gV, sV0);
  wait_async0();
  __syncthreads();

  for (int j0 = 0; j0 < Sc; j0 += 32) {
    const int cur = (j0 >> 5) & 1;
    if (j0 + 32 < Sc) {             // prefetch next key tile
      const int nxt = cur ^ 1;
      async_load_b128(gK + (size_t)(j0 + 32) * HDc, sK0 + nxt * KSL);
      async_load_b128(gV + (j0 + 32),               sV0 + nxt * VSL);
    }

    // scores: 2 q-subtiles x 2 key-subtiles, K-depth 64 -> 8 WMMAs
    const _Float16* Kt = Ks[cur];
    v16h kb[2][2];
    #pragma unroll
    for (int kj = 0; kj < 2; ++kj)
      #pragma unroll
      for (int kk = 0; kk < 2; ++kk)
        kb[kj][kk] = frag_b_load(Kt + (kj * 16) * 64 + kk * 32, 64, lane);

    v8f s[2][2] = {};
    #pragma unroll
    for (int qi = 0; qi < 2; ++qi)
      #pragma unroll
      for (int kj = 0; kj < 2; ++kj) {
        s[qi][kj] = WMMA_F16(qf[qi][0], kb[kj][0], s[qi][kj]);
        s[qi][kj] = WMMA_F16(qf[qi][1], kb[kj][1], s[qi][kj]);
      }

    // online softmax (rows reduced across each 16-lane half)
    #pragma unroll
    for (int qi = 0; qi < 2; ++qi) {
      #pragma unroll
      for (int r = 0; r < 8; ++r) {
        float a0 = s[qi][0][r], a1 = s[qi][1][r];
        float mx = fmaxf(a0, a1);
        #pragma unroll
        for (int d = 1; d < 16; d <<= 1) mx = fmaxf(mx, __shfl_xor(mx, d, 32));
        float mnew = fmaxf(mrow[qi][r], mx);
        float p0 = exp2f((a0 - mnew) * LOG2E);
        float p1 = exp2f((a1 - mnew) * LOG2E);
        float rs = p0 + p1;
        #pragma unroll
        for (int d = 1; d < 16; d <<= 1) rs += __shfl_xor(rs, d, 32);
        float scale = exp2f((mrow[qi][r] - mnew) * LOG2E);
        lrow[qi][r] = lrow[qi][r] * scale + rs;
        mrow[qi][r] = mnew;
        o[qi][0][r] *= scale; o[qi][1][r] *= scale;
        o[qi][2][r] *= scale; o[qi][3][r] *= scale;
        int prow = qi * 16 + r + hi;
        P[prow * 32 + ln]      = (_Float16)p0;
        P[prow * 32 + 16 + ln] = (_Float16)p1;
      }
    }

    // P(32x32) @ V(32x64): V fragments from LDS, shared across q-subtiles
    const _Float16* Vtl = Vs[cur];
    v16h vb[4];
    #pragma unroll
    for (int c = 0; c < 4; ++c)
      vb[c] = frag_b_load(Vtl + (c * 16) * 32, 32, lane);
    #pragma unroll
    for (int qi = 0; qi < 2; ++qi) {
      v16h pf = frag_a_load(P + qi * 16 * 32, 32, lane);
      #pragma unroll
      for (int c = 0; c < 4; ++c)
        o[qi][c] = WMMA_F16(pf, vb[c], o[qi][c]);
    }

    wait_async0();     // next tile landed (overlapped with work above)
    __syncthreads();   // all waves done reading cur
  }

  #pragma unroll
  for (int qi = 0; qi < 2; ++qi) {
    #pragma unroll
    for (int r = 0; r < 8; ++r) {
      int row = b * Sc + q0 + qi * 16 + r + hi;
      float inv = 1.0f / lrow[qi][r];
      _Float16* dst = Oout + (size_t)row * Dc + h * HDc + ln;
      dst[0]  = (_Float16)(o[qi][0][r] * inv);
      dst[16] = (_Float16)(o[qi][1][r] * inv);
      dst[32] = (_Float16)(o[qi][2][r] * inv);
      dst[48] = (_Float16)(o[qi][3][r] * inv);
    }
  }
}

// Output projection: attn[4096x1024] @ Wout^T + bias -> fp32 out.
__global__ __launch_bounds__(256) void out_gemm_kernel(
    const _Float16* __restrict__ X,
    const _Float16* __restrict__ W,
    const float* __restrict__ bias,
    float* __restrict__ Out) {
  __shared__ __align__(32) _Float16 As[2 * 128 * 32];
  __shared__ __align__(32) _Float16 Bs[2 * 128 * 32];

  const int lane = threadIdx.x & 31;
  const int wave = threadIdx.x >> 5;
  const int wm = wave >> 1, wn = wave & 1;
  const int nblk = Dc / 128;                     // 8
  const int mb = blockIdx.x / nblk, nb = blockIdx.x % nblk;
  const int m0 = mb * 128 + wm * 32;
  const int n0 = nb * 128 + wn * 64;

  v8f acc[2][4] = {};
  gemm_block_128x128<Dc>(X + (size_t)(mb * 128) * Dc, W + (size_t)(nb * 128) * Dc,
                         As, Bs, lane, wm, wn, acc);

  const int hi = (lane >> 4) << 3;
  const int ln = lane & 15;
  #pragma unroll
  for (int j = 0; j < 4; ++j) {
    float bv = bias[n0 + j * 16 + ln];
    #pragma unroll
    for (int i = 0; i < 2; ++i) {
      #pragma unroll
      for (int r = 0; r < 8; ++r) {
        Out[(size_t)(m0 + i * 16 + r + hi) * Dc + n0 + j * 16 + ln] =
            acc[i][j][r] + bv;
      }
    }
  }
}

// ---------------------------------------------------------------------------
extern "C" void kernel_launch(void* const* d_in, const int* in_sizes, int n_in,
                              void* d_out, int out_size, void* d_ws, size_t ws_size,
                              hipStream_t stream) {
  (void)in_sizes; (void)n_in; (void)out_size; (void)ws_size;

  const float* latents = (const float*)d_in[0];
  const float* w_qkv   = (const float*)d_in[1];
  const float* w_out   = (const float*)d_in[2];
  const float* b_out   = (const float*)d_in[3];
  float* out = (float*)d_out;

  char* ws = (char*)d_ws;
  size_t off = 0;
  auto salloc = [&](size_t bytes) -> void* {
    void* p = ws + off;
    off += (bytes + 255) & ~(size_t)255;
    return p;
  };
  _Float16* lat16  = (_Float16*)salloc(sizeof(_Float16) * Mrows * Dc);  //  8 MB
  _Float16* wqkv16 = (_Float16*)salloc(sizeof(_Float16) * Nqkv  * Dc);  //  6 MB
  _Float16* wout16 = (_Float16*)salloc(sizeof(_Float16) * Dc    * Dc);  //  2 MB
  _Float16* Qb     = (_Float16*)salloc(sizeof(_Float16) * Mrows * Dc);  //  8 MB
  _Float16* Kb     = (_Float16*)salloc(sizeof(_Float16) * Mrows * Dc);  //  8 MB
  _Float16* Vt     = (_Float16*)salloc(sizeof(_Float16) * Mrows * Dc);  //  8 MB
  _Float16* attn16 = (_Float16*)salloc(sizeof(_Float16) * Mrows * Dc);  //  8 MB

  f32_to_f16_kernel<<<512, 256, 0, stream>>>(latents, lat16, Mrows * Dc);
  f32_to_f16_kernel<<<512, 256, 0, stream>>>(w_qkv, wqkv16, Nqkv * Dc);
  f32_to_f16_kernel<<<512, 256, 0, stream>>>(w_out, wout16, Dc * Dc);

  // (4096/128) x (3072/128) = 32 x 24 = 768 blocks
  qkv_gemm_kernel<<<(Mrows / 128) * (Nqkv / 128), 256, 0, stream>>>(
      lat16, wqkv16, Qb, Kb, Vt);

  // 2*16*(2048/256) = 256 blocks; 8 waves x 32 queries each
  attn_kernel<<<Bc * Hc * (Sc / 256), 256, 0, stream>>>(Qb, Kb, Vt, attn16);

  // (4096/128) x (1024/128) = 32 x 8 = 256 blocks
  out_gemm_kernel<<<(Mrows / 128) * (Dc / 128), 256, 0, stream>>>(
      attn16, wout16, b_out, out);
}
